// modeler_78649441124468
// MI455X (gfx1250) — matrix operand order repaired and checked by reading, not compiled, not run on hardware
//
#include <hip/hip_runtime.h>

// ---------------- constants (match reference) ----------------
#define N_ALL   8000
#define NTYPE   4000      // NT
#define FTDIM   256
#define HID     256       // H1 == H2 == OUT
#define ATTD    64
#define NRELS   4
#define EREL    65536
#define EA      131072
#define BATCH   2048
#define NCLS    8

typedef __attribute__((ext_vector_type(16))) __bf16 v16bf;
typedef __attribute__((ext_vector_type(8)))  __bf16 v8bf;
typedef __attribute__((ext_vector_type(8)))  float  v8f;

// f32 -> bf16 bits with round-to-nearest-even
__device__ __forceinline__ unsigned short bfbits(float f) {
    union { float f; unsigned u; } in; in.f = f;
    unsigned u = in.u + 0x7FFFu + ((in.u >> 16) & 1u);
    return (unsigned short)(u >> 16);
}

// ---------------- utility: zero fill ----------------
__global__ void k_zero(float* __restrict__ p, size_t n) {
    size_t i = (size_t)blockIdx.x * blockDim.x + threadIdx.x;
    size_t stride = (size_t)gridDim.x * blockDim.x;
    for (; i < n; i += stride) p[i] = 0.0f;
}

// ---------------- edge scatter: msg[row] += w * src[col] ----------------
__global__ void k_edge_agg(const float* __restrict__ src_feat,
                           const int* __restrict__ rows,
                           const int* __restrict__ cols,
                           const float* __restrict__ w,
                           int dst_off,
                           float* __restrict__ msg) {
    int e = blockIdx.x;           // one edge per block
    int d = threadIdx.x;          // 256 dims
    int r = rows[e];
    int c = cols[e] + dst_off;
    float v = w[e] * src_feat[(size_t)c * FTDIM + d];
    atomicAdd(&msg[(size_t)r * FTDIM + d], v);
}

// ---------------- WMMA GEMM: C[M,N] = act(A[M,K] @ W[K,N] + bias) ----------------
// Block: 128 threads = 4 waves. Output tile 64x64 (each wave: 16 rows x 64 cols,
// i.e. 4 WMMAs per K-step). A and B tiles staged in LDS as bf16; B transposed so
// both fragment loads are contiguous 16B LDS reads matching ISA 7.12.2 layouts.
// Requires N%64==0, K%32==0; M is guarded.
__global__ __launch_bounds__(128)
void k_gemm_wmma(const float* __restrict__ A, int lda,
                 const float* __restrict__ W, int ldw,
                 const float* __restrict__ bias,
                 float* __restrict__ C, int ldc,
                 int M, int K, int do_relu) {
    __shared__ unsigned short lsA[64][32];   // [row][k]  bf16 bits
    __shared__ unsigned short lsB[64][32];   // [col][k]  bf16 bits (transposed tile)

    const int tid  = threadIdx.x;
    const int wave = tid >> 5;
    const int lane = tid & 31;
    const int half = lane >> 4;              // 0: lanes 0-15, 1: lanes 16-31
    const int lid  = lane & 15;

    const int mbase = blockIdx.x * 64;
    const int nbase = blockIdx.y * 64;

    v8f acc[4] = {};

    for (int kk = 0; kk < K; kk += 32) {
        // ---- stage A tile: 64 rows x 32 k  (512 float4-chunks / 128 threads) ----
#pragma unroll
        for (int c = 0; c < 4; ++c) {
            int idx = tid + c * 128;         // 0..511
            int row = idx >> 3;              // 0..63
            int k4  = (idx & 7) << 2;        // 0,4,...,28
            int gr  = mbase + row; if (gr >= M) gr = M - 1;
            const float4 v = *(const float4*)(A + (size_t)gr * lda + kk + k4);
            unsigned short* d = &lsA[row][k4];
            d[0] = bfbits(v.x); d[1] = bfbits(v.y); d[2] = bfbits(v.z); d[3] = bfbits(v.w);
            if (kk + 32 < K)
                __builtin_prefetch(A + (size_t)gr * lda + kk + 32 + k4, 0, 1);
        }
        // ---- stage B tile transposed: 32 k-rows x 64 cols ----
#pragma unroll
        for (int c = 0; c < 4; ++c) {
            int idx  = tid + c * 128;        // 0..511
            int krow = idx >> 4;             // 0..31
            int n4   = (idx & 15) << 2;      // 0,4,...,60
            const float4 v = *(const float4*)(W + (size_t)(kk + krow) * ldw + nbase + n4);
            lsB[n4 + 0][krow] = bfbits(v.x);
            lsB[n4 + 1][krow] = bfbits(v.y);
            lsB[n4 + 2][krow] = bfbits(v.z);
            lsB[n4 + 3][krow] = bfbits(v.w);
            if (kk + 32 < K)
                __builtin_prefetch(W + (size_t)(kk + 32 + krow) * ldw + nbase + n4, 0, 1);
        }
        __syncthreads();

        // A fragment (ISA 16-bit A layout): lane half selects K offset 0/8,
        // halves e0..7 -> K=kb+e, e8..15 -> K=kb+16+(e-8): two 16B chunks.
        const int arow = wave * 16 + lid;
        const v8bf a_lo = *(const v8bf*)&lsA[arow][half * 8];
        const v8bf a_hi = *(const v8bf*)&lsA[arow][half * 8 + 16];
        const v16bf afrag = __builtin_shufflevector(a_lo, a_hi,
            0, 1, 2, 3, 4, 5, 6, 7, 8, 9, 10, 11, 12, 13, 14, 15);

#pragma unroll
        for (int nt = 0; nt < 4; ++nt) {
            // B fragment: same K layout as A (lane half selects K offset 0/8, chunks at +0/+16).
            const int bcol = nt * 16 + lid;
            const v8bf b_lo = *(const v8bf*)&lsB[bcol][half * 8];
            const v8bf b_hi = *(const v8bf*)&lsB[bcol][half * 8 + 16];
            const v16bf bfrag = __builtin_shufflevector(b_lo, b_hi,
                0, 1, 2, 3, 4, 5, 6, 7, 8, 9, 10, 11, 12, 13, 14, 15);
            acc[nt] = __builtin_amdgcn_wmma_f32_16x16x32_bf16(
                          false, afrag, false, bfrag, (short)0, acc[nt], false, false);
        }
        __syncthreads();
    }

    // ---- epilogue: bias + optional relu ----
#pragma unroll
    for (int nt = 0; nt < 4; ++nt) {
        const int col = nbase + nt * 16 + lid;
        const float bc = bias ? bias[col] : 0.0f;
#pragma unroll
        for (int j = 0; j < 8; ++j) {
            const int row = mbase + wave * 16 + j + half * 8;  // C/D: VGPR j -> M=j(+8)
            if (row < M) {
                float v = acc[nt][j] + bc;
                if (do_relu) v = fmaxf(v, 0.0f);
                C[(size_t)row * ldc + col] = v;
            }
        }
    }
}

// ---------------- semantic attention: accumulate sum_n w[p,n] ----------------
// wave per (node, p); lane a handles att dims a and a+32
__global__ __launch_bounds__(32)
void k_att_w(const float* __restrict__ v0, const float* __restrict__ v1,
             const float* __restrict__ W,   // [HID, ATTD]
             const float* __restrict__ bvec,// [ATTD]
             const float* __restrict__ q,   // [ATTD]
             float* __restrict__ wsum) {    // [2]
    int node = blockIdx.x;
    int p    = blockIdx.y;
    const float* v = (p == 0 ? v0 : v1) + (size_t)node * HID;
    int lane = threadIdx.x;
    float acc0 = bvec[lane];
    float acc1 = bvec[lane + 32];
    for (int d = 0; d < HID; ++d) {
        float x = v[d];
        acc0 += x * W[d * ATTD + lane];
        acc1 += x * W[d * ATTD + lane + 32];
    }
    float wv = tanhf(acc0) * q[lane] + tanhf(acc1) * q[lane + 32];
    for (int off = 16; off > 0; off >>= 1) wv += __shfl_down(wv, off, 32);
    if (lane == 0) atomicAdd(&wsum[p], wv);
}

__global__ void k_beta(const float* __restrict__ wsum, float* __restrict__ beta, float invn) {
    float m0 = wsum[0] * invn, m1 = wsum[1] * invn;
    float mx = fmaxf(m0, m1);
    float e0 = expf(m0 - mx), e1 = expf(m1 - mx);
    float inv = 1.0f / (e0 + e1);
    beta[0] = e0 * inv; beta[1] = e1 * inv;
}

__global__ void k_combine(const float* __restrict__ v0, const float* __restrict__ v1,
                          const float* __restrict__ beta, float* __restrict__ out) {
    size_t i = (size_t)blockIdx.x * blockDim.x + threadIdx.x;   // NTYPE*HID threads
    out[i] = beta[0] * v0[i] + beta[1] * v1[i];
}

// ---------------- vcat = [o1[n] | features_n] ----------------
__global__ void k_vcat(const float* __restrict__ o1n, const float* __restrict__ featn,
                       float* __restrict__ vcat) {
    size_t i = (size_t)blockIdx.x * blockDim.x + threadIdx.x;   // NTYPE*512
    int row = (int)(i >> 9);
    int col = (int)(i & 511);
    vcat[i] = (col < HID) ? o1n[(size_t)row * HID + col]
                          : featn[(size_t)row * FTDIM + (col - HID)];
}

// ---------------- per-edge scalar MLP + segment-sum into node_struct ----------------
// 4 waves per block, wave per edge
__global__ __launch_bounds__(128)
void k_edge_mlp(const float* __restrict__ A_val, const int* __restrict__ A_type,
                const int* __restrict__ A_col,
                const float* __restrict__ feW1, const float* __restrict__ feb1,
                const float* __restrict__ feW2, const float* __restrict__ feb2,
                float* __restrict__ node_struct) {
    int e = blockIdx.x * 4 + (threadIdx.x >> 5);
    int lane = threadIdx.x & 31;
    if (e >= EA) return;
    float ew = A_val[e];
    int t = A_type[e];
    const float* W1 = feW1 + t * HID;
    const float* b1 = feb1 + t * HID;
    const float* W2 = feW2 + t * HID;
    float acc = 0.0f;
    for (int d = lane; d < HID; d += 32) {
        float h = fmaxf(ew * W1[d] + b1[d], 0.0f);
        acc += h * W2[d];
    }
    for (int off = 16; off > 0; off >>= 1) acc += __shfl_down(acc, off, 32);
    if (lane == 0) atomicAdd(&node_struct[A_col[e]], acc + feb2[t]);
}

// ---------------- node scalar MLP: s[i] ----------------
__global__ void k_node_s(const float* __restrict__ ns,
                         const float* __restrict__ W1, const float* __restrict__ b1,
                         const float* __restrict__ W2, const float* __restrict__ b2,
                         float* __restrict__ s) {
    int i = blockIdx.x * blockDim.x + threadIdx.x;
    if (i >= N_ALL) return;
    float x = ns[i];
    float acc = b2[0];
    for (int d = 0; d < HID; ++d) acc += fmaxf(x * W1[d] + b1[d], 0.0f) * W2[d];
    s[i] = acc;
}

// ---------------- triple dot-products + g MLPs ----------------
__global__ __launch_bounds__(256)
void k_triple(const float* __restrict__ A, const float* __restrict__ A2,
              const float* __restrict__ s, const int* __restrict__ train_x,
              const float* __restrict__ gW1, const float* __restrict__ gb1,
              const float* __restrict__ gW2, const float* __restrict__ gb2,
              float* __restrict__ out_struct) {
    int b   = blockIdx.x;
    int tid = threadIdx.x;
    int src = train_x[b * 3 + 0];
    int mid = train_x[b * 3 + 1];
    int dst = train_x[b * 3 + 2];
    const float* As = A  + (size_t)src * N_ALL;  const float* A2s = A2 + (size_t)src * N_ALL;
    const float* Am = A  + (size_t)mid * N_ALL;  const float* A2m = A2 + (size_t)mid * N_ALL;
    const float* Ad = A  + (size_t)dst * N_ALL;  const float* A2d = A2 + (size_t)dst * N_ALL;
    float t1 = 0.f, t2 = 0.f, t3 = 0.f;
    for (int j = tid; j < N_ALL; j += 256) {
        float sj = s[j];
        float ms = (As[j] + 0.8f * A2s[j]) * sj;
        float mm = (Am[j] + 0.8f * A2m[j]) * sj;
        float md = (Ad[j] + 0.8f * A2d[j]) * sj;
        t1 += ms * mm; t2 += ms * md; t3 += mm * md;
    }
    __shared__ float r1[256], r2[256], r3[256];
    r1[tid] = t1; r2[tid] = t2; r3[tid] = t3;
    __syncthreads();
    for (int st = 128; st > 0; st >>= 1) {
        if (tid < st) { r1[tid] += r1[tid + st]; r2[tid] += r2[tid + st]; r3[tid] += r3[tid + st]; }
        __syncthreads();
    }
    if (tid < 3) {
        float t = (tid == 0) ? r1[0] : (tid == 1) ? r2[0] : r3[0];
        float acc = gb2[0];
        for (int d = 0; d < HID; ++d) acc += fmaxf(t * gW1[d] + gb1[d], 0.0f) * gW2[d];
        out_struct[b * 3 + tid] = acc;
    }
}

// ---------------- hsum = embs2[src]+embs2[mid]+embs2[dst] ----------------
__global__ void k_hsum(const float* __restrict__ embs2, const int* __restrict__ train_x,
                       float* __restrict__ hsum) {
    size_t i = (size_t)blockIdx.x * blockDim.x + threadIdx.x;   // BATCH*HID
    int b = (int)(i >> 8);
    int d = (int)(i & 255);
    int src = train_x[b * 3 + 0];
    int mid = train_x[b * 3 + 1];
    int dst = train_x[b * 3 + 2];
    hsum[i] = embs2[(size_t)src * HID + d] + embs2[(size_t)mid * HID + d]
            + embs2[(size_t)dst * HID + d];
}

// ---------------- logits = [h2 | out_struct] @ mlp_W2 + b2 ----------------
__global__ void k_logits(const float* __restrict__ h2, const float* __restrict__ ostr,
                         const float* __restrict__ W2, const float* __restrict__ b2,
                         float* __restrict__ logits) {
    int i = blockIdx.x * blockDim.x + threadIdx.x;   // BATCH*NCLS
    if (i >= BATCH * NCLS) return;
    int b = i >> 3, c = i & 7;
    float acc = b2[c];
    const float* h = h2 + (size_t)b * 128;
    for (int j = 0; j < 128; ++j) acc += h[j] * W2[j * NCLS + c];
    for (int j = 0; j < 3; ++j)   acc += ostr[b * 3 + j] * W2[(128 + j) * NCLS + c];
    logits[i] = acc;
}

// =============================================================================
extern "C" void kernel_launch(void* const* d_in, const int* in_sizes, int n_in,
                              void* d_out, int out_size, void* d_ws, size_t ws_size,
                              hipStream_t stream) {
    (void)in_sizes; (void)n_in; (void)out_size; (void)ws_size;
    const float* features = (const float*)d_in[0];
    const float* rel_w    = (const float*)d_in[1];
    const float* A_val    = (const float*)d_in[2];
    const float* A_dense  = (const float*)d_in[3];
    const float* A2_dense = (const float*)d_in[4];
    const float* gcn0_W   = (const float*)d_in[5];
    const float* gcn0_b   = (const float*)d_in[6];
    const float* gcn1_W   = (const float*)d_in[7];
    const float* gcn1_b   = (const float*)d_in[8];
    const float* satt0_W  = (const float*)d_in[9];
    const float* satt0_b  = (const float*)d_in[10];
    const float* satt0_q  = (const float*)d_in[11];
    const float* satt1_W  = (const float*)d_in[12];
    const float* satt1_b  = (const float*)d_in[13];
    const float* satt1_q  = (const float*)d_in[14];
    const float* fc_W     = (const float*)d_in[15];
    const float* fc_b     = (const float*)d_in[16];
    const float* fe_W1    = (const float*)d_in[17];
    const float* fe_b1    = (const float*)d_in[18];
    const float* fe_W2    = (const float*)d_in[19];
    const float* fe_b2    = (const float*)d_in[20];
    const float* fn_W1    = (const float*)d_in[21];
    const float* fn_b1    = (const float*)d_in[22];
    const float* fn_W2    = (const float*)d_in[23];
    const float* fn_b2    = (const float*)d_in[24];
    const float* g_W1     = (const float*)d_in[25];
    const float* g_b1     = (const float*)d_in[26];
    const float* g_W2     = (const float*)d_in[27];
    const float* g_b2     = (const float*)d_in[28];
    const float* mlp_W1   = (const float*)d_in[29];
    const float* mlp_b1   = (const float*)d_in[30];
    const float* mlp_W2   = (const float*)d_in[31];
    const float* mlp_b2   = (const float*)d_in[32];
    const int* rel_rows   = (const int*)d_in[33];
    const int* rel_cols   = (const int*)d_in[34];
    const int* A_col      = (const int*)d_in[35];
    const int* A_type     = (const int*)d_in[36];
    const int* train_x    = (const int*)d_in[37];

    float* embs2  = (float*)d_out;                        // [8000,256]
    float* logits = (float*)d_out + (size_t)N_ALL * HID;  // [2048,8]

    // ---- workspace layout ----
    float* ws = (float*)d_ws;
    size_t off = 0;
    float* msg   = ws + off; off += (size_t)NRELS * NTYPE * HID;   // 4.10 M
    float* vecb  = ws + off; off += (size_t)NRELS * NTYPE * HID;   // 4.10 M
    float* embs1 = ws + off; off += (size_t)N_ALL * HID;           // 2.05 M
    float* o1b   = ws + off; off += (size_t)N_ALL * HID;           // 2.05 M
    float* vcat  = ws + off; off += (size_t)NTYPE * 512;           // 2.05 M
    float* wsum  = ws + off; off += 2;
    float* beta  = ws + off; off += 2;
    float* nstr  = ws + off; off += N_ALL;
    float* sbuf  = ws + off; off += N_ALL;
    float* ostr  = ws + off; off += (size_t)BATCH * 3;
    float* hsum  = ws + off; off += (size_t)BATCH * HID;
    float* h2    = ws + off; off += (size_t)BATCH * 128;

    const int REL_DST[4] = {1, 0, 0, 1};
    const int MT_NT = (NTYPE + 63) / 64;    // 63 M-tiles for M=4000

    auto hetero_layer = [&](const float* src_feat,
                            const float* gW, const float* gb,
                            const float* sW, const float* sb, const float* sq,
                            float* out8000) {
        k_zero<<<1024, 256, 0, stream>>>(msg, (size_t)NRELS * NTYPE * HID);
        for (int r = 0; r < NRELS; ++r) {
            k_edge_agg<<<EREL, 256, 0, stream>>>(
                src_feat, rel_rows + (size_t)r * EREL, rel_cols + (size_t)r * EREL,
                rel_w + (size_t)r * EREL, REL_DST[r] * NTYPE, msg + (size_t)r * NTYPE * HID);
            k_gemm_wmma<<<dim3(MT_NT, HID / 64), 128, 0, stream>>>(
                msg + (size_t)r * NTYPE * HID, HID,
                gW + (size_t)r * HID * HID, HID,
                gb + (size_t)r * HID,
                vecb + (size_t)r * NTYPE * HID, HID, NTYPE, HID, 1);
        }
        for (int n = 0; n < 2; ++n) {
            const float* v0 = vecb + (size_t)(2 * n) * NTYPE * HID;
            const float* v1 = vecb + (size_t)(2 * n + 1) * NTYPE * HID;
            k_zero<<<1, 2, 0, stream>>>(wsum, 2);
            k_att_w<<<dim3(NTYPE, 2), 32, 0, stream>>>(
                v0, v1, sW + (size_t)n * HID * ATTD, sb + (size_t)n * ATTD,
                sq + (size_t)n * ATTD, wsum);
            k_beta<<<1, 1, 0, stream>>>(wsum, beta, 1.0f / NTYPE);
            k_combine<<<(NTYPE * HID) / 256, 256, 0, stream>>>(
                v0, v1, beta, out8000 + (size_t)n * NTYPE * HID);
        }
    };

    // ---- two hetero GCN layers ----
    hetero_layer(features, gcn0_W, gcn0_b, satt0_W, satt0_b, satt0_q, embs1);
    hetero_layer(embs1,    gcn1_W, gcn1_b, satt1_W, satt1_b, satt1_q, o1b);

    // ---- fc: embs2 = [o1 | features] @ fc_W + fc_b ----
    for (int n = 0; n < 2; ++n) {
        k_vcat<<<(NTYPE * 512) / 256, 256, 0, stream>>>(
            o1b + (size_t)n * NTYPE * HID, features + (size_t)n * NTYPE * FTDIM, vcat);
        k_gemm_wmma<<<dim3(MT_NT, HID / 64), 128, 0, stream>>>(
            vcat, 512, fc_W + (size_t)n * 512 * HID, HID, fc_b + (size_t)n * HID,
            embs2 + (size_t)n * NTYPE * HID, HID, NTYPE, 512, 0);
    }

    // ---- structural scalar path ----
    k_zero<<<64, 256, 0, stream>>>(nstr, N_ALL);
    k_edge_mlp<<<EA / 4, 128, 0, stream>>>(A_val, A_type, A_col,
                                           fe_W1, fe_b1, fe_W2, fe_b2, nstr);
    k_node_s<<<(N_ALL + 255) / 256, 256, 0, stream>>>(nstr, fn_W1, fn_b1, fn_W2, fn_b2, sbuf);
    k_triple<<<BATCH, 256, 0, stream>>>(A_dense, A2_dense, sbuf, train_x,
                                        g_W1, g_b1, g_W2, g_b2, ostr);

    // ---- final MLP ----
    k_hsum<<<(BATCH * HID) / 256, 256, 0, stream>>>(embs2, train_x, hsum);
    k_gemm_wmma<<<dim3(BATCH / 64, 128 / 64), 128, 0, stream>>>(
        hsum, HID, mlp_W1, 128, mlp_b1, h2, 128, BATCH, HID, 1);
    k_logits<<<(BATCH * NCLS + 255) / 256, 256, 0, stream>>>(h2, ostr, mlp_W2, mlp_b2, logits);
}
